// ObeliskLayer_18545668784748
// MI455X (gfx1250) — matrix operand order repaired and checked by reading, not compiled, not run on hardware
//
#include <hip/hip_runtime.h>
#include <hip/hip_bf16.h>

typedef __bf16 bf16;
typedef __attribute__((ext_vector_type(16))) __bf16 v16bf;
typedef __attribute__((ext_vector_type(8)))  __bf16 v8bf;
typedef __attribute__((ext_vector_type(4)))  __bf16 v4bf;
typedef __attribute__((ext_vector_type(8)))  float  v8f;
typedef __attribute__((ext_vector_type(4)))  unsigned int v4u;
typedef __attribute__((ext_vector_type(8)))  int v8i;
typedef __attribute__((ext_vector_type(4)))  int v4i;

#define NB    16      // batch
#define NPTS  4096    // H/4 * W/4
#define KPTS  128     // sample points
#define CHMAX 640     // final channel count
#define EPS   1e-5f

// ---------------------------------------------------------------- WMMA helpers
__device__ __forceinline__ v8f wmma_bf16(v16bf a, v16bf b, v8f c) {
  return __builtin_amdgcn_wmma_f32_16x16x32_bf16(false, a, false, b, (short)0, c,
                                                 false, false);
}

// A-frag (16x32 bf16): lane m = lane%16; elem e -> K = kb + half*16 + e
__device__ __forceinline__ v16bf load_a_bf16(const bf16* row, int kb, int half) {
  v8bf lo = *(const v8bf*)(row + kb + half * 16);
  v8bf hi = *(const v8bf*)(row + kb + half * 16 + 8);
  v16bf f;
#pragma unroll
  for (int e = 0; e < 8; ++e) { f[e] = lo[e]; f[e + 8] = hi[e]; }
  return f;
}
__device__ __forceinline__ v16bf load_a_f32(const float* row, int kb, int half) {
  const float* p0 = row + kb + half * 16;
  const float* p1 = row + kb + half * 16 + 8;
  v16bf f;
#pragma unroll
  for (int e = 0; e < 8; ++e) { f[e] = (__bf16)p0[e]; f[e + 8] = (__bf16)p1[e]; }
  return f;
}
// B-frag (32x16 bf16): lane n = lane%16; elem e -> K = kb + half*16 + e
__device__ __forceinline__ v16bf load_b_bf16(const bf16* row, int kb, int half) {
  v8bf lo = *(const v8bf*)(row + kb + half * 16);
  v8bf hi = *(const v8bf*)(row + kb + half * 16 + 8);
  v16bf f;
#pragma unroll
  for (int e = 0; e < 8; ++e) { f[e] = lo[e]; f[e + 8] = hi[e]; }
  return f;
}

// ---------------------------------------------------------------- 0) w1 -> bf16
__global__ void __launch_bounds__(256)
convertw_kernel(const float* __restrict__ w1, bf16* __restrict__ Wb) {
  int i = blockIdx.x * 256 + threadIdx.x;       // 16384 threads, 4 elems each
  float4 f = ((const float4*)w1)[i];
  v4bf o; o[0] = (__bf16)f.x; o[1] = (__bf16)f.y; o[2] = (__bf16)f.z; o[3] = (__bf16)f.w;
  *(v4bf*)(Wb + 4 * i) = o;
}

// ---------------------------------------------------------------- 1) sampler
// S[b][c][n][k] bf16  (k contiguous => coalesced stores; x is L2-resident)
__global__ void __launch_bounds__(256)
sample_kernel(const float* __restrict__ x, const float* __restrict__ off,
              bf16* __restrict__ S) {
  unsigned idx = blockIdx.x * 256u + threadIdx.x;   // 2^23 threads
  int k = idx & 127;
  int n = (idx >> 7) & 4095;
  int b = idx >> 19;
  int py = n >> 6, px = n & 63;
  float gx = 2.0f * (px + 0.5f) * (1.0f / 64.0f) - 1.0f + off[2 * k + 0];
  float gy = 2.0f * (py + 0.5f) * (1.0f / 64.0f) - 1.0f + off[2 * k + 1];
  float fx = ((gx + 1.0f) * 256.0f - 1.0f) * 0.5f;
  float fy = ((gy + 1.0f) * 256.0f - 1.0f) * 0.5f;
  float x0f = floorf(fx), y0f = floorf(fy);
  float wx = fx - x0f, wy = fy - y0f;
  int x0 = (int)x0f, y0 = (int)y0f, x1 = x0 + 1, y1 = y0 + 1;
  float vx0 = (x0 >= 0 && x0 < 256) ? 1.f : 0.f;
  float vx1 = (x1 >= 0 && x1 < 256) ? 1.f : 0.f;
  float vy0 = (y0 >= 0 && y0 < 256) ? 1.f : 0.f;
  float vy1 = (y1 >= 0 && y1 < 256) ? 1.f : 0.f;
  int cx0 = x0 < 0 ? 0 : (x0 > 255 ? 255 : x0);
  int cx1 = x1 < 0 ? 0 : (x1 > 255 ? 255 : x1);
  int cy0 = y0 < 0 ? 0 : (y0 > 255 ? 255 : y0);
  int cy1 = y1 < 0 ? 0 : (y1 > 255 ? 255 : y1);
  float w00 = (1.f - wx) * (1.f - wy) * vx0 * vy0;
  float w10 = wx * (1.f - wy) * vx1 * vy0;
  float w01 = (1.f - wx) * wy * vx0 * vy1;
  float w11 = wx * wy * vx1 * vy1;
#pragma unroll
  for (int c = 0; c < 4; ++c) {
    const float* P = x + ((size_t)(b * 4 + c) << 16);
    float v = w00 * P[cy0 * 256 + cx0] + w10 * P[cy0 * 256 + cx1] +
              w01 * P[cy1 * 256 + cx0] + w11 * P[cy1 * 256 + cx1];
    S[(((size_t)(b * 4 + c) * NPTS + n) << 7) + k] = (__bf16)v;
  }
}

// ---------------------------------------------------------------- 2) grouped GEMM
// per (b,g): out(128 x 4096) = w1[g](128x128) @ S[b][g](128 x 4096), relu
// w1[g] tile staged into LDS by the Tensor Data Mover (32 KB, 1-D descriptor).
__global__ void __launch_bounds__(256)
gemm1_kernel(const bf16* __restrict__ S, const bf16* __restrict__ Wb,
             bf16* __restrict__ H) {
  __shared__ __align__(16) bf16 As[128 * 128];   // 32 KB: w1[g] bf16
  int blk = blockIdx.x;                          // B*G*32
  int ntile = blk & 31;
  int g = (blk >> 5) & 3;
  int b = blk >> 7;

  int wave = threadIdx.x >> 5, lane = threadIdx.x & 31;
  if (wave == 0) {
    // Tensor DMA: 1-D tensor, 16384 bf16 elements (32 KB), global -> LDS.
    unsigned lds_addr = (unsigned)(uintptr_t)(&As[0]);
    unsigned long long ga = (unsigned long long)(uintptr_t)(Wb + (size_t)g * 16384);
    // D# group0: count=1 | lds_addr | global_addr[56:0] | type=2
    v4u g0 = {1u, lds_addr, (unsigned)(ga & 0xFFFFFFFFu),
              (unsigned)((ga >> 32) & 0x1FFFFFFu) | (2u << 30)};
    // D# group1: data_size=2B (code 1); tensor_dim0=16384; tensor_dim1=1;
    // tile_dim0=16384; tile_dim1=1; strides=16384
    v8i g1 = {(int)(1u << 16),                    // workgroup_mask=0, data_size=1
              (int)((16384u & 0xFFFFu) << 16),    // tensor_dim0 lo16 in [63:48]
              (int)(1u << 16),                    // dim0 hi16=0, tensor_dim1 lo16=1
              (int)(16384u << 16),                // dim1 hi16=0, tile_dim0=16384
              1,                                  // tile_dim1=1, tile_dim2=0
              16384,                              // tensor_dim0_stride lo32
              (int)(16384u << 16),                // stride0 hi16=0, stride1 lo16
              0};                                 // tensor_dim1_stride hi32
    v4i g2 = {1, 1, 0, 0};                        // tensor_dim2=1, tensor_dim3=1
    v4i g3 = {0, 0, 0, 0};
    v8i g4 = {0, 0, 0, 0, 0, 0, 0, 0};            // unused trailing group
    __builtin_amdgcn_tensor_load_to_lds(g0, g1, g2, g3, g4, 0);
    __builtin_amdgcn_s_wait_tensorcnt(0);         // TDM done before barrier
  }
  __syncthreads();

  int half = lane >> 4, nl = lane & 15;
  int n = ntile * 128 + wave * 16 + nl;
  const bf16* Brow = S + ((size_t)(b * 4 + g) * NPTS + n) * 128;

  v8f acc[8];
#pragma unroll
  for (int t = 0; t < 8; ++t) acc[t] = (v8f){0, 0, 0, 0, 0, 0, 0, 0};

#pragma unroll
  for (int kb = 0; kb < 128; kb += 32) {
    v16bf bf = load_b_bf16(Brow, kb, half);
#pragma unroll
    for (int t = 0; t < 8; ++t) {
      v16bf af = load_a_bf16(As + (size_t)(t * 16 + nl) * 128, kb, half);
      acc[t] = wmma_bf16(af, bf, acc[t]);
    }
  }

  bf16* Hrow = H + ((size_t)b * NPTS + n) * CHMAX + g * 128;
#pragma unroll
  for (int t = 0; t < 8; ++t) {
    v8bf o;
#pragma unroll
    for (int r = 0; r < 8; ++r) o[r] = (__bf16)fmaxf(acc[t][r], 0.0f);
    *(v8bf*)(Hrow + t * 16 + 8 * half) = o;                 // 8 consecutive ch
  }
}

// ---------------------------------------------------------------- 3) instance-norm stats
__global__ void __launch_bounds__(256)
stats_kernel(const bf16* __restrict__ H, float* __restrict__ stats,
             int ch_off, int nch) {
  __shared__ float ssum[2048];
  __shared__ float ssq[2048];
  int chunks = nch >> 3;
  int chunk = blockIdx.x % chunks;
  int b = blockIdx.x / chunks;
  int ch0 = ch_off + chunk * 8;
  int tid = threadIdx.x;
  float sum[8] = {0, 0, 0, 0, 0, 0, 0, 0}, sq[8] = {0, 0, 0, 0, 0, 0, 0, 0};
  for (int n = tid; n < NPTS; n += 256) {
    v8bf v = *(const v8bf*)(H + ((size_t)b * NPTS + n) * CHMAX + ch0);
#pragma unroll
    for (int j = 0; j < 8; ++j) { float f = (float)v[j]; sum[j] += f; sq[j] += f * f; }
  }
#pragma unroll
  for (int j = 0; j < 8; ++j) { ssum[tid * 8 + j] = sum[j]; ssq[tid * 8 + j] = sq[j]; }
  __syncthreads();
  for (int s = 128; s > 0; s >>= 1) {
    if (tid < s) {
#pragma unroll
      for (int j = 0; j < 8; ++j) {
        ssum[tid * 8 + j] += ssum[(tid + s) * 8 + j];
        ssq[tid * 8 + j] += ssq[(tid + s) * 8 + j];
      }
    }
    __syncthreads();
  }
  if (tid < 8) {
    float mean = ssum[tid] * (1.0f / NPTS);
    float var = ssq[tid] * (1.0f / NPTS) - mean * mean;
    float* sp = stats + ((size_t)b * CHMAX + ch0 + tid) * 2;
    sp[0] = mean;
    sp[1] = rsqrtf(var + EPS);
  }
}

// ---------------------------------------------------------------- 4) dense layer GEMM
// y(32 x 4096) = relu( wd(32 x cin) @ inorm(H[0:cin]) ), appended at ch=co_off.
__global__ void __launch_bounds__(256)
dense_kernel(const bf16* __restrict__ H, const float* __restrict__ stats,
             const float* __restrict__ Wd, int cin, int co_off,
             bf16* __restrict__ Hout) {
  __shared__ float s_mean[CHMAX];
  __shared__ float s_rstd[CHMAX];
  int nblk = blockIdx.x & 31;
  int b = blockIdx.x >> 5;
  const float* st = stats + (size_t)b * CHMAX * 2;
  for (int i = threadIdx.x; i < cin; i += 256) { s_mean[i] = st[2 * i]; s_rstd[i] = st[2 * i + 1]; }
  __syncthreads();

  int wave = threadIdx.x >> 5, lane = threadIdx.x & 31;
  int half = lane >> 4, nl = lane & 15;
  int n = nblk * 128 + wave * 16 + nl;
  const bf16* Hrow = H + ((size_t)b * NPTS + n) * CHMAX;

  v8f acc0 = (v8f){0, 0, 0, 0, 0, 0, 0, 0};
  v8f acc1 = (v8f){0, 0, 0, 0, 0, 0, 0, 0};
  for (int kb = 0; kb < cin; kb += 32) {
    if (kb + 32 < cin)
      __builtin_prefetch(Hrow + kb + 32 + half * 16, 0, 0);  // global_prefetch_b8
    v16bf bf;
    {
      const bf16* hp = Hrow + kb + half * 16;
      v8bf lo = *(const v8bf*)hp;
      v8bf hi = *(const v8bf*)(hp + 8);
#pragma unroll
      for (int e = 0; e < 8; ++e) {
        int ch = kb + half * 16 + e;
        bf[e]     = (__bf16)(((float)lo[e] - s_mean[ch]) * s_rstd[ch]);
        bf[e + 8] = (__bf16)(((float)hi[e] - s_mean[ch + 8]) * s_rstd[ch + 8]);
      }
    }
    v16bf a0 = load_a_f32(Wd + (size_t)nl * cin, kb, half);
    v16bf a1 = load_a_f32(Wd + (size_t)(16 + nl) * cin, kb, half);
    acc0 = wmma_bf16(a0, bf, acc0);
    acc1 = wmma_bf16(a1, bf, acc1);
  }

  bf16* Orow = Hout + ((size_t)b * NPTS + n) * CHMAX + co_off;
  v8bf o;
#pragma unroll
  for (int r = 0; r < 8; ++r) o[r] = (__bf16)fmaxf(acc0[r], 0.0f);
  *(v8bf*)(Orow + 8 * half) = o;
#pragma unroll
  for (int r = 0; r < 8; ++r) o[r] = (__bf16)fmaxf(acc1[r], 0.0f);
  *(v8bf*)(Orow + 16 + 8 * half) = o;
}

// ---------------------------------------------------------------- 5) final projection
// F[b][co][n] f32 = relu( w3(64x640) @ inorm(H[0:640]) )
__global__ void __launch_bounds__(256)
final_kernel(const bf16* __restrict__ H, const float* __restrict__ stats,
             const float* __restrict__ W3, float* __restrict__ F) {
  __shared__ float s_mean[CHMAX];
  __shared__ float s_rstd[CHMAX];
  int nblk = blockIdx.x & 31;
  int b = blockIdx.x >> 5;
  const float* st = stats + (size_t)b * CHMAX * 2;
  for (int i = threadIdx.x; i < CHMAX; i += 256) { s_mean[i] = st[2 * i]; s_rstd[i] = st[2 * i + 1]; }
  __syncthreads();

  int wave = threadIdx.x >> 5, lane = threadIdx.x & 31;
  int half = lane >> 4, nl = lane & 15;
  int n = nblk * 128 + wave * 16 + nl;
  const bf16* Hrow = H + ((size_t)b * NPTS + n) * CHMAX;

  v8f acc[4];
#pragma unroll
  for (int t = 0; t < 4; ++t) acc[t] = (v8f){0, 0, 0, 0, 0, 0, 0, 0};

  for (int kb = 0; kb < CHMAX; kb += 32) {
    v16bf bf;
    {
      const bf16* hp = Hrow + kb + half * 16;
      v8bf lo = *(const v8bf*)hp;
      v8bf hi = *(const v8bf*)(hp + 8);
#pragma unroll
      for (int e = 0; e < 8; ++e) {
        int ch = kb + half * 16 + e;
        bf[e]     = (__bf16)(((float)lo[e] - s_mean[ch]) * s_rstd[ch]);
        bf[e + 8] = (__bf16)(((float)hi[e] - s_mean[ch + 8]) * s_rstd[ch + 8]);
      }
    }
#pragma unroll
    for (int t = 0; t < 4; ++t) {
      v16bf af = load_a_f32(W3 + (size_t)(t * 16 + nl) * CHMAX, kb, half);
      acc[t] = wmma_bf16(af, bf, acc[t]);
    }
  }
#pragma unroll
  for (int t = 0; t < 4; ++t)
#pragma unroll
    for (int r = 0; r < 8; ++r) {
      int ch = t * 16 + 8 * half + r;
      F[((size_t)(b * 64 + ch)) * NPTS + n] = fmaxf(acc[t][r], 0.0f);
    }
}

// ---------------------------------------------------------------- 6) 2x bilinear upsample
__global__ void __launch_bounds__(256)
upsample_kernel(const float* __restrict__ F, float* __restrict__ out) {
  unsigned idx = blockIdx.x * 256u + threadIdx.x;  // 2^24 = 16*64*128*128
  int ox = idx & 127;
  int oy = (idx >> 7) & 127;
  int bc = idx >> 14;
  float sx = (ox + 0.5f) * 0.5f - 0.5f;
  float sy = (oy + 0.5f) * 0.5f - 0.5f;
  float fx = floorf(sx), fy = floorf(sy);
  float wx = sx - fx, wy = sy - fy;
  int x0 = (int)fx, y0 = (int)fy;
  int x0c = x0 < 0 ? 0 : (x0 > 63 ? 63 : x0);
  int x1c = (x0 + 1) < 0 ? 0 : ((x0 + 1) > 63 ? 63 : (x0 + 1));
  int y0c = y0 < 0 ? 0 : (y0 > 63 ? 63 : y0);
  int y1c = (y0 + 1) < 0 ? 0 : ((y0 + 1) > 63 ? 63 : (y0 + 1));
  const float* P = F + (size_t)bc * NPTS;
  float v = (1.f - wx) * (1.f - wy) * P[y0c * 64 + x0c] +
            wx * (1.f - wy) * P[y0c * 64 + x1c] +
            (1.f - wx) * wy * P[y1c * 64 + x0c] +
            wx * wy * P[y1c * 64 + x1c];
  out[idx] = v;
}

// ---------------------------------------------------------------- launch
extern "C" void kernel_launch(void* const* d_in, const int* in_sizes, int n_in,
                              void* d_out, int out_size, void* d_ws, size_t ws_size,
                              hipStream_t stream) {
  const float* x   = (const float*)d_in[0];
  const float* off = (const float*)d_in[1];
  const float* w1  = (const float*)d_in[2];
  const float* wd[4] = {(const float*)d_in[3], (const float*)d_in[4],
                        (const float*)d_in[5], (const float*)d_in[6]};
  const float* w3  = (const float*)d_in[7];
  float* out = (float*)d_out;

  char* ws = (char*)d_ws;
  const size_t S_BYTES  = (size_t)NB * 4 * NPTS * KPTS * sizeof(bf16);   // 64 MiB
  const size_t H_BYTES  = (size_t)NB * NPTS * CHMAX * sizeof(bf16);      // 80 MiB
  const size_t ST_BYTES = (size_t)NB * CHMAX * 2 * sizeof(float);        // 80 KiB
  const size_t F_BYTES  = (size_t)NB * 64 * NPTS * sizeof(float);        // 16 MiB
  bf16*  S     = (bf16*)ws;
  bf16*  H     = (bf16*)(ws + S_BYTES);
  float* stats = (float*)(ws + S_BYTES + H_BYTES);
  float* F     = (float*)(ws + S_BYTES + H_BYTES + ST_BYTES);
  bf16*  Wb    = (bf16*)(ws + S_BYTES + H_BYTES + ST_BYTES + F_BYTES);   // 128 KiB

  convertw_kernel<<<64, 256, 0, stream>>>(w1, Wb);
  sample_kernel<<<(NB * NPTS * KPTS) / 256, 256, 0, stream>>>(x, off, S);
  gemm1_kernel<<<NB * 4 * 32, 256, 0, stream>>>(S, Wb, H);
  stats_kernel<<<NB * (512 / 8), 256, 0, stream>>>(H, stats, 0, 512);
  int cin = 512;
  for (int i = 0; i < 4; ++i) {
    dense_kernel<<<NB * 32, 256, 0, stream>>>(H, stats, wd[i], cin, cin, H);
    stats_kernel<<<NB * (32 / 8), 256, 0, stream>>>(H, stats, cin, 32);
    cin += 32;
  }
  final_kernel<<<NB * 32, 256, 0, stream>>>(H, stats, w3, F);
  upsample_kernel<<<(NB * 64 * 128 * 128) / 256, 256, 0, stream>>>(F, out);
}
